// InvariantPointAttention_62612033241667
// MI455X (gfx1250) — matrix operand order repaired and checked by reading, not compiled, not run on hardware
//
#include <hip/hip_runtime.h>
#include <hip/hip_bf16.h>
#include <math.h>

// ---------------------------------------------------------------------------
// IPA for MI455X (gfx1250, wave32).
//  - bf16 WMMA (16x16x32) for all big GEMMs, fp32 accumulate
//  - f32 WMMA (16x16x4) for attention logits (K=16, reference precision)
//  - TDM tensor_load_to_lds for the z@Wb stream (TENSORcnt path)
//  - global_load_async_to_lds_b128 for the o_pair z stream (ASYNCcnt path)
// ---------------------------------------------------------------------------

#define USE_TDM   1
#define USE_ASYNC 1

typedef __attribute__((ext_vector_type(16))) __bf16 bf16x16;
typedef __attribute__((ext_vector_type(8)))  float  f32x8;
typedef __attribute__((ext_vector_type(2)))  float  f32x2;
typedef __attribute__((ext_vector_type(8)))  unsigned int u32x8;
typedef __attribute__((ext_vector_type(4)))  unsigned int u32x4;
typedef __attribute__((ext_vector_type(8)))  int i32x8;
typedef __attribute__((ext_vector_type(4)))  int i32x4;

#define NRES 512
#define CS   384
#define CZ   128
#define CH   16
#define NH   12
#define PQN  4
#define PVN  8

__device__ __forceinline__ __bf16 f2bf(float x) { return (__bf16)x; }

__device__ __forceinline__ f32x8 wmma_bf16(bf16x16 a, bf16x16 b, f32x8 c) {
  return __builtin_amdgcn_wmma_f32_16x16x32_bf16(false, a, false, b,
                                                 (short)0, c, false, false);
}
__device__ __forceinline__ f32x8 wmma_f32k4(f32x2 a, f32x2 b, f32x8 c) {
  return __builtin_amdgcn_wmma_f32_16x16x4_f32(false, a, false, b,
                                               (short)0, c, false, false);
}

// A fragment 16x32 bf16 row-major (ld even, base 4B aligned): packed u32 loads.
__device__ __forceinline__ bf16x16 load_A(const __bf16* p, int ld, int lane) {
  int m = lane & 15, g = lane >> 4;
  const unsigned int* q = (const unsigned int*)(p + (size_t)m * ld);
  u32x8 a;
#pragma unroll
  for (int v = 0; v < 8; ++v) {
    int k = g * 16 + 2 * v;
    a[v] = q[k >> 1];
  }
  return __builtin_bit_cast(bf16x16, a);
}

// A fragment from fp32 panel (convert on load).
__device__ __forceinline__ bf16x16 load_A_f32(const float* p, int ld, int lane) {
  int m = lane & 15, g = lane >> 4;
  bf16x16 a;
#pragma unroll
  for (int v = 0; v < 8; ++v) {
    int k = g * 16 + 2 * v;
    a[2 * v]     = f2bf(p[(size_t)m * ld + k]);
    a[2 * v + 1] = f2bf(p[(size_t)m * ld + k + 1]);
  }
  return a;
}

// B fragment from pair-interleaved LDS: u32 word [kpair*16 + n] = {B[2kp][n],B[2kp+1][n]}
__device__ __forceinline__ bf16x16 load_Bp(const unsigned int* p, int lane) {
  int n = lane & 15, g = lane >> 4;
  u32x8 b;
#pragma unroll
  for (int v = 0; v < 8; ++v) b[v] = p[(g * 8 + v) * 16 + n];
  return __builtin_bit_cast(bf16x16, b);
}

// B fragment 32x16 from bf16 K-major (global), per-element.
__device__ __forceinline__ bf16x16 load_B(const __bf16* p, int ld, int lane) {
  int n = lane & 15, g = lane >> 4;
  bf16x16 b;
#pragma unroll
  for (int v = 0; v < 8; ++v) {
    int k = g * 16 + 2 * v;
    b[2 * v]     = p[(size_t)k * ld + n];
    b[2 * v + 1] = p[(size_t)(k + 1) * ld + n];
  }
  return b;
}

// B fragment 32x16 from fp32 K-major (LDS), convert on load.
__device__ __forceinline__ bf16x16 load_B_f32(const float* p, int ld, int lane) {
  int n = lane & 15, g = lane >> 4;
  bf16x16 b;
#pragma unroll
  for (int v = 0; v < 8; ++v) {
    int k = g * 16 + 2 * v;
    b[2 * v]     = f2bf(p[(size_t)k * ld + n]);
    b[2 * v + 1] = f2bf(p[(size_t)(k + 1) * ld + n]);
  }
  return b;
}

#if USE_ASYNC
__device__ __forceinline__ void async_load_b128(void* lds_dst, const void* gsrc) {
  unsigned int lds = (unsigned int)(size_t)lds_dst;  // addrspace(3) offset
  asm volatile("global_load_async_to_lds_b128 %0, %1, off"
               :: "v"(lds), "v"(gsrc) : "memory");
}
__device__ __forceinline__ void wait_async0() {
  asm volatile("s_wait_asynccnt 0" ::: "memory");
}
#endif

#if USE_TDM && defined(__HIP_DEVICE_COMPILE__) && __has_builtin(__builtin_amdgcn_tensor_load_to_lds)
#define HAVE_TDM 1
// 1-D contiguous DMA of nelem fp32 into LDS via Tensor Data Mover.
__device__ __forceinline__ void tdm_load_1d(void* lds_dst, const void* gsrc, unsigned nelem) {
  unsigned long long ga = (unsigned long long)gsrc;
  u32x4 g0;
  g0[0] = 1u;                                                // count=1, user mode
  g0[1] = (unsigned int)(size_t)lds_dst;                     // lds_addr
  g0[2] = (unsigned int)ga;                                  // global_addr[31:0]
  g0[3] = (unsigned int)((ga >> 32) & 0x1FFFFFFu) | (2u << 30);  // addr[56:32] | type=2
  i32x8 g1;
  g1[0] = (2 << 16);                       // data_size = 4B
  g1[1] = (int)((nelem & 0xFFFFu) << 16);  // tensor_dim0[15:0]  (bits 63:48)
  g1[2] = (int)((nelem >> 16) & 0xFFFFu) | (1 << 16);  // dim0[31:16] | tensor_dim1=1
  g1[3] = (int)((nelem & 0xFFFFu) << 16);  // tile_dim0 (bits 127:112)
  g1[4] = 1;                               // tile_dim1 = 1
  g1[5] = (int)nelem;                      // tensor_dim0_stride[31:0]
  g1[6] = 0;
  g1[7] = 0;
  i32x4 gz = {0, 0, 0, 0};
#if __clang_major__ >= 23
  i32x8 gz8 = {0, 0, 0, 0, 0, 0, 0, 0};
  __builtin_amdgcn_tensor_load_to_lds(g0, g1, gz, gz, gz8, 0);
#else
  __builtin_amdgcn_tensor_load_to_lds(g0, g1, gz, gz, 0);
#endif
}
#else
#define HAVE_TDM 0
#endif

// ---------------------------------------------------------------------------
// Setup kernels
// ---------------------------------------------------------------------------
__global__ void cvt_f32_bf16(const float* __restrict__ src, __bf16* __restrict__ dst, int n) {
  int i = blockIdx.x * blockDim.x + threadIdx.x;
  if (i < n) dst[i] = f2bf(src[i]);
}

__global__ void build_wb(const float* __restrict__ Wb, __bf16* __restrict__ Wb_bf) {
  int idx = blockIdx.x * blockDim.x + threadIdx.x;  // 128*16
  if (idx < 128 * 16) {
    int k = idx / 16, h = idx % 16;
    Wb_bf[idx] = f2bf(h < NH ? Wb[k * NH + h] : 0.f);
  }
}

__global__ void build_wfull(const float* __restrict__ Whid, const float* __restrict__ Wpair,
                            const float* __restrict__ Wpts, const float* __restrict__ Wpn,
                            const float* __restrict__ bhid, const float* __restrict__ bpair,
                            const float* __restrict__ bpts, const float* __restrict__ bpn,
                            __bf16* __restrict__ Wfull, float* __restrict__ bias_sum) {
  int idx = blockIdx.x * blockDim.x + threadIdx.x;  // 2112*384
  if (idx < 2112 * 384) {
    int r = idx / 384, c = idx % 384;
    float v;
    if (r < 192)       v = Whid[r * 384 + c];
    else if (r < 1728) v = Wpair[(r - 192) * 384 + c];
    else if (r < 2016) v = Wpts[(r - 1728) * 384 + c];
    else               v = Wpn[(r - 2016) * 384 + c];
    Wfull[idx] = f2bf(v);
  }
  if (idx < 384) bias_sum[idx] = bhid[idx] + bpair[idx] + bpts[idx] + bpn[idx];
}

// ---------------------------------------------------------------------------
// Generic bf16 GEMM: C(MxN f32) = A(MxK bf16) @ B(KxN bf16) + bias. 1 wave/tile.
// ---------------------------------------------------------------------------
__global__ void __launch_bounds__(32)
gemm_bf16_k(const __bf16* __restrict__ A, const __bf16* __restrict__ B,
            const float* __restrict__ bias, float* __restrict__ C,
            int M, int N, int K) {
  __shared__ __align__(16) __bf16 sA[16 * 32];
  __shared__ __align__(16) __bf16 sB[32 * 16];  // pair-interleaved
  int lane = threadIdx.x;
  int tm = blockIdx.x, tn = blockIdx.y;
  const __bf16* Ap = A + (size_t)tm * 16 * K;
  const __bf16* Bp = B + tn * 16;
  f32x8 acc = {};
  for (int k0 = 0; k0 < K; k0 += 32) {
    if (k0 + 32 < K)
      __builtin_prefetch(Ap + (size_t)(lane & 15) * K + k0 + 32, 0, 3);
#pragma unroll
    for (int t = 0; t < 16; ++t) {
      int e = t * 32 + lane;
      sA[e] = Ap[(size_t)(e >> 5) * K + k0 + (e & 31)];
    }
#pragma unroll
    for (int t = 0; t < 16; ++t) {
      int e = t * 32 + lane;
      int krow = e >> 4, n = e & 15;
      sB[(krow >> 1) * 32 + n * 2 + (krow & 1)] = Bp[(size_t)(k0 + krow) * N + n];
    }
    __syncthreads();
    acc = wmma_bf16(load_A(sA, 32, lane), load_Bp((const unsigned int*)sB, lane), acc);
    __syncthreads();
  }
  int n = tn * 16 + (lane & 15);
  float bv = bias ? bias[n] : 0.f;
#pragma unroll
  for (int r = 0; r < 8; ++r) {
    int m = tm * 16 + (lane >> 4) * 8 + r;
    C[(size_t)m * N + n] = acc[r] + bv;
  }
}

// ---------------------------------------------------------------------------
// biasz[h][i][j] = (z @ Wb)[ij,h] + bb[h].  z panel (16 rows x 128) DMA'd by TDM.
// ---------------------------------------------------------------------------
__global__ void __launch_bounds__(32)
bias_z_k(const float* __restrict__ z, const __bf16* __restrict__ Wb_bf,
         const float* __restrict__ bb, float* __restrict__ biasz) {
  __shared__ __align__(16) float  sZ[16 * CZ];   // 8KB fp32 panel
  __shared__ __align__(16) __bf16 sB[CZ * 16];   // Wb, pair-interleaved (all K)
  int lane = threadIdx.x;
  int tm = blockIdx.x;  // 16384 tiles of 16 flattened (i,j) rows
  const float* Zp = z + (size_t)tm * 16 * CZ;
#if HAVE_TDM
  tdm_load_1d(sZ, Zp, 16 * CZ);                  // TENSORcnt path
#else
#pragma unroll
  for (int t = 0; t < 64; ++t) { int e = t * 32 + lane; sZ[e] = Zp[e]; }
#endif
  // stage Wb while DMA is in flight
#pragma unroll
  for (int t = 0; t < 64; ++t) {
    int e = t * 32 + lane;
    int krow = e >> 4, n = e & 15;
    sB[(krow >> 1) * 32 + n * 2 + (krow & 1)] = Wb_bf[krow * 16 + n];
  }
#if HAVE_TDM
  __builtin_amdgcn_s_wait_tensorcnt((short)0);
#endif
  __syncthreads();
  f32x8 acc = {};
#pragma unroll
  for (int k0 = 0; k0 < CZ; k0 += 32) {
    acc = wmma_bf16(load_A_f32(sZ + k0, CZ, lane),
                    load_Bp((const unsigned int*)sB + (k0 >> 1) * 16, lane), acc);
  }
  int h = lane & 15;
  if (h < NH) {
    float bv = bb[h];
#pragma unroll
    for (int r = 0; r < 8; ++r) {
      int mrow = tm * 16 + (lane >> 4) * 8 + r;  // = i*512 + j
      int i = mrow >> 9, j = mrow & 511;
      biasz[((size_t)h * NRES + i) * NRES + j] = acc[r] + bv;
    }
  }
}

// ---------------------------------------------------------------------------
// Rotary + rigid-frame point transforms; pack operands. One thread per (i,h).
// ---------------------------------------------------------------------------
__global__ void fixup_k(const float* __restrict__ qlin, const float* __restrict__ kvlin,
                        const float* __restrict__ qplin, const float* __restrict__ kvplin,
                        const float* __restrict__ rot, const float* __restrict__ trans,
                        float* __restrict__ qf32, float* __restrict__ kTf32,
                        __bf16* __restrict__ v_bf, __bf16* __restrict__ vpts_bf,
                        float* __restrict__ qpts, float* __restrict__ kpts) {
  int idx = blockIdx.x * blockDim.x + threadIdx.x;
  if (idx >= NRES * NH) return;
  int i = idx / NH, h = idx % NH;

  float qv[CH], kvv[CH], vv[CH];
#pragma unroll
  for (int c = 0; c < CH; ++c) {
    qv[c]  = qlin[(size_t)i * (NH * CH) + h * CH + c];
    kvv[c] = kvlin[(size_t)i * (NH * 2 * CH) + h * 2 * CH + c];
    vv[c]  = kvlin[(size_t)i * (NH * 2 * CH) + h * 2 * CH + CH + c];
  }
#pragma unroll
  for (int f = 0; f < 4; ++f) {  // rotary over first 8 dims
    float infq = powf(10000.f, -(float)f * 0.25f);
    float ang = (float)i * infq;
    float c = __cosf(ang), s = __sinf(ang);
    float x1 = qv[2 * f], x2 = qv[2 * f + 1];
    qv[2 * f] = x1 * c - x2 * s; qv[2 * f + 1] = x1 * s + x2 * c;
    x1 = kvv[2 * f]; x2 = kvv[2 * f + 1];
    kvv[2 * f] = x1 * c - x2 * s; kvv[2 * f + 1] = x1 * s + x2 * c;
  }
#pragma unroll
  for (int c = 0; c < CH; ++c) {
    qf32[((size_t)h * NRES + i) * CH + c] = qv[c];
    kTf32[((size_t)h * CH + c) * NRES + i] = kvv[c];  // k^T (K-major) for logits B
    v_bf[((size_t)h * NRES + i) * CH + c] = f2bf(vv[c]);
  }

  const float* R = rot + (size_t)i * 9;
  float T0 = trans[i * 3], T1 = trans[i * 3 + 1], T2 = trans[i * 3 + 2];
#pragma unroll
  for (int p = 0; p < PQN; ++p) {
    float a0 = qplin[(size_t)i * 144 + 0 * 48 + h * PQN + p];
    float a1 = qplin[(size_t)i * 144 + 1 * 48 + h * PQN + p];
    float a2 = qplin[(size_t)i * 144 + 2 * 48 + h * PQN + p];
    size_t base = ((size_t)i * NH + h) * 12 + p * 3;
    qpts[base + 0] = R[0] * a0 + R[1] * a1 + R[2] * a2 + T0;
    qpts[base + 1] = R[3] * a0 + R[4] * a1 + R[5] * a2 + T1;
    qpts[base + 2] = R[6] * a0 + R[7] * a1 + R[8] * a2 + T2;
  }
#pragma unroll
  for (int p = 0; p < PQN + PVN; ++p) {
    float a0 = kvplin[(size_t)i * 432 + 0 * 144 + h * 12 + p];
    float a1 = kvplin[(size_t)i * 432 + 1 * 144 + h * 12 + p];
    float a2 = kvplin[(size_t)i * 432 + 2 * 144 + h * 12 + p];
    float x = R[0] * a0 + R[1] * a1 + R[2] * a2 + T0;
    float y = R[3] * a0 + R[4] * a1 + R[5] * a2 + T1;
    float zc = R[6] * a0 + R[7] * a1 + R[8] * a2 + T2;
    if (p < PQN) {
      size_t base = ((size_t)i * NH + h) * 12 + p * 3;
      kpts[base + 0] = x; kpts[base + 1] = y; kpts[base + 2] = zc;
    } else {
      int pv = p - PQN;
      size_t base = ((size_t)h * NRES + i) * 32 + pv * 3;
      vpts_bf[base + 0] = f2bf(x); vpts_bf[base + 1] = f2bf(y); vpts_bf[base + 2] = f2bf(zc);
    }
  }
#pragma unroll
  for (int c = 24; c < 32; ++c)
    vpts_bf[((size_t)h * NRES + i) * 32 + c] = (__bf16)0.f;
}

// ---------------------------------------------------------------------------
// Attention logits: fp32 WMMA (4 x K=4) + scaled pair bias + point term + mask.
// ---------------------------------------------------------------------------
__global__ void __launch_bounds__(32)
logits_k(const float* __restrict__ qf32, const float* __restrict__ kTf32,
         const float* __restrict__ biasz, const float* __restrict__ qpts,
         const float* __restrict__ kpts, const float* __restrict__ head_weights,
         const float* __restrict__ mask, float* __restrict__ aout) {
  int h = blockIdx.x, tm = blockIdx.y, tn = blockIdx.z, lane = threadIdx.x;
  int m = lane & 15, g = lane >> 4;
  const float* qrow = qf32 + ((size_t)h * NRES + tm * 16 + m) * CH;
  const float* kcol = kTf32 + (size_t)h * CH * NRES + tn * 16 + (lane & 15);
  f32x8 acc = {};
#pragma unroll
  for (int kk = 0; kk < CH; kk += 4) {
    f32x2 af; af[0] = qrow[kk + 2 * g];               af[1] = qrow[kk + 2 * g + 1];
    f32x2 bf; bf[0] = kcol[(size_t)(kk + 2 * g) * NRES]; bf[1] = kcol[(size_t)(kk + 2 * g + 1) * NRES];
    acc = wmma_f32k4(af, bf, acc);
  }

  const float s_qk = 0.14433756729740643f;  // sqrt(1/(3*16))
  const float s_b  = 0.57735026918962584f;  // sqrt(1/3)
  float hw = logf(1.f + __expf(head_weights[h])) * 0.13608276348795434f;  // softplus*sqrt(1/54)

  int j = tn * 16 + (lane & 15);
  float kb[12];
  const float* kbp = kpts + ((size_t)j * NH + h) * 12;
#pragma unroll
  for (int t = 0; t < 12; ++t) kb[t] = kbp[t];
  float mj = mask[j];
#pragma unroll
  for (int r = 0; r < 8; ++r) {
    int i = tm * 16 + (lane >> 4) * 8 + r;
    const float* qb = qpts + ((size_t)i * NH + h) * 12;
    float d2 = 0.f;
#pragma unroll
    for (int t = 0; t < 12; ++t) { float d = qb[t] - kb[t]; d2 += d * d; }
    float val = acc[r] * s_qk + s_b * biasz[((size_t)h * NRES + i) * NRES + j]
              - 0.5f * hw * d2 + 100000.f * (mask[i] * mj - 1.f);
    aout[((size_t)h * NRES + i) * NRES + j] = val;
  }
}

// Row softmax over j (512), one block per (h,i) row, in place.
__global__ void __launch_bounds__(256)
softmax_k(float* __restrict__ a) {
  __shared__ float red[256];
  float* p = a + (size_t)blockIdx.x * NRES;
  int tid = threadIdx.x;
  float x0 = p[tid], x1 = p[tid + 256];
  red[tid] = fmaxf(x0, x1);
  __syncthreads();
  for (int s = 128; s > 0; s >>= 1) { if (tid < s) red[tid] = fmaxf(red[tid], red[tid + s]); __syncthreads(); }
  float m = red[0];
  __syncthreads();
  float e0 = __expf(x0 - m), e1 = __expf(x1 - m);
  red[tid] = e0 + e1;
  __syncthreads();
  for (int s = 128; s > 0; s >>= 1) { if (tid < s) red[tid] += red[tid + s]; __syncthreads(); }
  float inv = 1.f / red[0];
  p[tid] = e0 * inv;
  p[tid + 256] = e1 * inv;
}

// ---------------------------------------------------------------------------
// o = a@v, o_pt = a@v_pts : per (h, i-tile), 3 waves share the A tile.
// ---------------------------------------------------------------------------
__global__ void __launch_bounds__(96)
attn_out_k(const float* __restrict__ a, const __bf16* __restrict__ v_bf,
           const __bf16* __restrict__ vpts_bf,
           float* __restrict__ o_raw, float* __restrict__ optraw) {
  __shared__ __align__(16) __bf16 sA[16 * 32];
  int h = blockIdx.x, tm = blockIdx.y;
  int tid = threadIdx.x, wave = tid >> 5, lane = tid & 31;
  const float* Ap = a + ((size_t)h * NRES + tm * 16) * NRES;
  f32x8 acc = {};
  for (int k0 = 0; k0 < NRES; k0 += 32) {
    for (int e = tid; e < 512; e += 96)
      sA[e] = f2bf(Ap[(size_t)(e >> 5) * NRES + k0 + (e & 31)]);
    __syncthreads();
    bf16x16 Af = load_A(sA, 32, lane);
    bf16x16 Bf;
    if (wave == 0) Bf = load_B(v_bf + ((size_t)h * NRES + k0) * CH, CH, lane);
    else           Bf = load_B(vpts_bf + ((size_t)h * NRES + k0) * 32 + (wave - 1) * 16, 32, lane);
    acc = wmma_bf16(Af, Bf, acc);
    __syncthreads();
  }
  int n = lane & 15;
#pragma unroll
  for (int r = 0; r < 8; ++r) {
    int i = tm * 16 + (lane >> 4) * 8 + r;
    if (wave == 0) {
      o_raw[((size_t)i * NH + h) * CH + n] = acc[r];
    } else {
      int col = (wave - 1) * 16 + n;
      if (col < 24) optraw[((size_t)i * NH + h) * 24 + col] = acc[r];
    }
  }
}

// ---------------------------------------------------------------------------
// o_pair[i] = a[:,i,:] @ z[i]  (M=12 pad16, K=512, N=128). z tile via async-to-LDS.
// ---------------------------------------------------------------------------
__global__ void __launch_bounds__(256)
opair_k(const float* __restrict__ a, const float* __restrict__ z,
        float* __restrict__ opair) {
  __shared__ __align__(16) __bf16 sA[16 * 32];
  __shared__ __align__(16) float  sZ[32 * CZ];   // 16KB fp32
  int i = blockIdx.x;
  int tid = threadIdx.x, wave = tid >> 5, lane = tid & 31;
  const float* Zp = z + (size_t)i * NRES * CZ;
  f32x8 acc = {};
  for (int k0 = 0; k0 < NRES; k0 += 32) {
    for (int e = tid; e < 512; e += 256) {
      int hh = e >> 5;
      sA[e] = f2bf(hh < NH ? a[((size_t)hh * NRES + i) * NRES + k0 + (e & 31)] : 0.f);
    }
#if USE_ASYNC
#pragma unroll
    for (int t = 0; t < 4; ++t) {              // 32x128 f32 = 1024 x 16B chunks
      int e = tid + t * 256;
      async_load_b128(sZ + e * 4, Zp + (size_t)k0 * CZ + e * 4);
    }
    wait_async0();
#else
    for (int e = tid; e < 32 * CZ; e += 256) sZ[e] = Zp[(size_t)k0 * CZ + e];
#endif
    __syncthreads();
    acc = wmma_bf16(load_A(sA, 32, lane), load_B_f32(sZ + wave * 16, CZ, lane), acc);
    __syncthreads();
  }
  int n = lane & 15;
#pragma unroll
  for (int r = 0; r < 8; ++r) {
    int h = (lane >> 4) * 8 + r;
    if (h < NH) opair[((size_t)i * NH + h) * CZ + wave * 16 + n] = acc[r];
  }
}

// ---------------------------------------------------------------------------
// Assemble 2112-wide bf16 feature: [o(192) | o_pair(1536) | pts_feat(288) | norms(96)]
// ---------------------------------------------------------------------------
__global__ void __launch_bounds__(256)
feat_k(const float* __restrict__ o_raw, const float* __restrict__ opair,
       const float* __restrict__ optraw, const float* __restrict__ rot,
       const float* __restrict__ trans, __bf16* __restrict__ feat) {
  int i = blockIdx.x, tid = threadIdx.x;
  __bf16* f = feat + (size_t)i * 2112;
  if (tid < 96) {
    int h = tid >> 3, p = tid & 7;
    const float* R = rot + (size_t)i * 9;
    const float* g = optraw + ((size_t)i * NH + h) * 24 + p * 3;
    float g0 = g[0] - trans[i * 3], g1 = g[1] - trans[i * 3 + 1], g2 = g[2] - trans[i * 3 + 2];
    float lx = R[0] * g0 + R[3] * g1 + R[6] * g2;   // R^T * g
    float ly = R[1] * g0 + R[4] * g1 + R[7] * g2;
    float lz = R[2] * g0 + R[5] * g1 + R[8] * g2;
    f[1728 + 0 * 96 + tid] = f2bf(lx);
    f[1728 + 1 * 96 + tid] = f2bf(ly);
    f[1728 + 2 * 96 + tid] = f2bf(lz);
    f[2016 + tid] = f2bf(sqrtf(lx * lx + ly * ly + lz * lz + 1e-8f));
  }
  for (int e = tid; e < 1728; e += 256) {
    float v = (e < 192) ? o_raw[(size_t)i * 192 + e] : opair[(size_t)i * 1536 + (e - 192)];
    f[e] = f2bf(v);
  }
}

// ---------------------------------------------------------------------------
// Host launch
// ---------------------------------------------------------------------------
extern "C" void kernel_launch(void* const* d_in, const int* in_sizes, int n_in,
                              void* d_out, int out_size, void* d_ws, size_t ws_size,
                              hipStream_t stream) {
  const float* s    = (const float*)d_in[0];
  const float* z    = (const float*)d_in[1];
  const float* rot  = (const float*)d_in[2];
  const float* trn  = (const float*)d_in[3];
  const float* mask = (const float*)d_in[4];
  const float* Wq   = (const float*)d_in[5];
  const float* bq   = (const float*)d_in[6];
  const float* Wkv  = (const float*)d_in[7];
  const float* bkv  = (const float*)d_in[8];
  const float* Wqp  = (const float*)d_in[9];
  const float* bqp  = (const float*)d_in[10];
  const float* Wkvp = (const float*)d_in[11];
  const float* bkvp = (const float*)d_in[12];
  const float* Wb   = (const float*)d_in[13];
  const float* bb   = (const float*)d_in[14];
  const float* hwts = (const float*)d_in[15];
  const float* Whid = (const float*)d_in[16];
  const float* bhid = (const float*)d_in[17];
  const float* Wpair= (const float*)d_in[18];
  const float* bpair= (const float*)d_in[19];
  const float* Wpts = (const float*)d_in[20];
  const float* bpts = (const float*)d_in[21];
  const float* Wpn  = (const float*)d_in[22];
  const float* bpn  = (const float*)d_in[23];

  float* s_out = (float*)d_out;              // 512*384
  float* aout  = (float*)d_out + NRES * CS;  // 12*512*512

  // workspace layout (bytes), ~25 MB total
  char* w = (char*)d_ws;
  __bf16* S_BF    = (__bf16*)(w + 0);         // 512x384
  __bf16* WQ_BF   = (__bf16*)(w + 393216);    // 384x192
  __bf16* WKV_BF  = (__bf16*)(w + 540672);    // 384x384
  __bf16* WQP_BF  = (__bf16*)(w + 835584);    // 384x144
  __bf16* WKVP_BF = (__bf16*)(w + 946176);    // 384x432
  __bf16* WB_BF   = (__bf16*)(w + 1277952);   // 128x16
  __bf16* WFULL   = (__bf16*)(w + 1282048);   // 2112x384
  float*  BSUM    = (float*)(w + 2904064);    // 384
  float*  QLIN    = (float*)(w + 2905600);    // 512x192
  float*  KVLIN   = (float*)(w + 3298816);    // 512x384
  float*  QPLIN   = (float*)(w + 4085248);    // 512x144
  float*  KVPLIN  = (float*)(w + 4380160);    // 512x432
  float*  QF32    = (float*)(w + 5264896);    // 12x512x16 f32
  float*  KTF32   = (float*)(w + 5658112);    // 12x16x512 f32
  __bf16* V_BF    = (__bf16*)(w + 6051328);   // 12x512x16
  __bf16* VPTS_BF = (__bf16*)(w + 6247936);   // 12x512x32
  float*  QPTS    = (float*)(w + 6641152);    // 512x12x12
  float*  KPTS    = (float*)(w + 6936064);    // 512x12x12
  float*  BIASZ   = (float*)(w + 7230976);    // 12x512x512
  float*  ORAW    = (float*)(w + 19813888);   // 512x12x16
  float*  OPTRAW  = (float*)(w + 20207104);   // 512x12x24
  float*  OPAIR   = (float*)(w + 20796928);   // 512x12x128
  __bf16* FEAT    = (__bf16*)(w + 23942656);  // 512x2112

  auto cvt = [&](const float* src, __bf16* dst, int n) {
    cvt_f32_bf16<<<(n + 255) / 256, 256, 0, stream>>>(src, dst, n);
  };
  cvt(s, S_BF, NRES * CS);
  cvt(Wq, WQ_BF, CS * 192);
  cvt(Wkv, WKV_BF, CS * 384);
  cvt(Wqp, WQP_BF, CS * 144);
  cvt(Wkvp, WKVP_BF, CS * 432);
  build_wb<<<(2048 + 255) / 256, 256, 0, stream>>>(Wb, WB_BF);
  build_wfull<<<(2112 * 384 + 255) / 256, 256, 0, stream>>>(
      Whid, Wpair, Wpts, Wpn, bhid, bpair, bpts, bpn, WFULL, BSUM);

  // projections (bf16 WMMA)
  gemm_bf16_k<<<dim3(32, 12), 32, 0, stream>>>(S_BF, WQ_BF, bq, QLIN, NRES, 192, CS);
  gemm_bf16_k<<<dim3(32, 24), 32, 0, stream>>>(S_BF, WKV_BF, bkv, KVLIN, NRES, 384, CS);
  gemm_bf16_k<<<dim3(32, 9), 32, 0, stream>>>(S_BF, WQP_BF, bqp, QPLIN, NRES, 144, CS);
  gemm_bf16_k<<<dim3(32, 27), 32, 0, stream>>>(S_BF, WKVP_BF, bkvp, KVPLIN, NRES, 432, CS);

  // pair bias z@Wb + bb (TDM + WMMA; streams 128MB of z)
  bias_z_k<<<16384, 32, 0, stream>>>(z, WB_BF, bb, BIASZ);

  // rotary + point transforms + operand packing
  fixup_k<<<(NRES * NH + 127) / 128, 128, 0, stream>>>(
      QLIN, KVLIN, QPLIN, KVPLIN, rot, trn, QF32, KTF32, V_BF, VPTS_BF, QPTS, KPTS);

  // logits (f32 WMMA qk + bias + point term + mask) -> a region of d_out
  logits_k<<<dim3(NH, 32, 32), 32, 0, stream>>>(QF32, KTF32, BIASZ, QPTS, KPTS, hwts, mask, aout);
  softmax_k<<<NH * NRES, 256, 0, stream>>>(aout);

  // o, o_pt (bf16 WMMA)
  attn_out_k<<<dim3(NH, 32), 96, 0, stream>>>(aout, V_BF, VPTS_BF, ORAW, OPTRAW);
  // o_pair (async-to-LDS + bf16 WMMA; streams z again)
  opair_k<<<NRES, 256, 0, stream>>>(aout, z, OPAIR);

  // feature assembly + fused final projection (bf16 WMMA, K=2112)
  feat_k<<<NRES, 256, 0, stream>>>(ORAW, OPAIR, OPTRAW, rot, trn, FEAT);
  gemm_bf16_k<<<dim3(32, 24), 32, 0, stream>>>(FEAT, WFULL, BSUM, s_out, NRES, CS, 2112);
  (void)in_sizes; (void)n_in; (void)out_size; (void)ws_size;
}
